// MultiHeadAttention_9491877724818
// MI455X (gfx1250) — matrix operand order repaired and checked by reading, not compiled, not run on hardware
//
#include <hip/hip_runtime.h>

// ---------------------------------------------------------------------------
// MI455X (gfx1250, wave32) multi-head attention block.
// All five GEMM stages run on v_wmma_f32_16x16x32_bf16 with f32 accumulation.
// GEMM staging uses CDNA5 async global->LDS DMA (ASYNCcnt) with double-
// buffered LDS so the copy of tile i+1 overlaps the WMMAs of tile i.
// Each wave register-blocks 32(M) x 64(N): 8 WMMAs per K-step, B fragments
// reused across the two M sub-tiles.
// ---------------------------------------------------------------------------

typedef __bf16 bf16x16 __attribute__((ext_vector_type(16)));
typedef float  f32x8   __attribute__((ext_vector_type(8)));
typedef unsigned int u32x4 __attribute__((ext_vector_type(4)));

union FragU { u32x4 q[2]; bf16x16 v; };

__device__ __forceinline__ unsigned short f2bf(float f) {
    union { float f; unsigned int u; } c; c.f = f;
    unsigned int u = c.u;
    u += 0x7fffu + ((u >> 16) & 1u);          // round-to-nearest-even
    return (unsigned short)(u >> 16);
}

// LDS byte offset of a generic pointer that points into shared memory.
__device__ __forceinline__ unsigned lds_off(const void* p) {
    return (unsigned)(unsigned long long)
        (__attribute__((address_space(3))) const void*)p;
}
// CDNA5 async DMA: 16B per lane, global -> LDS, tracked by ASYNCcnt.
__device__ __forceinline__ void async_load_b128(unsigned ldsDst, const void* gsrc) {
    asm volatile("global_load_async_to_lds_b128 %0, %1, off"
                 :: "v"(ldsDst), "v"(gsrc) : "memory");
}
__device__ __forceinline__ void wait_async0() {
    asm volatile("s_wait_asynccnt 0x0" ::: "memory");
}

// ---------------------------------------------------------------------------
// Elementwise f32 -> bf16 convert (weights keep their (N,K) layout).
// ---------------------------------------------------------------------------
__global__ void conv_f32_to_bf16(const float* __restrict__ src,
                                 unsigned short* __restrict__ dst,
                                 long long total) {
    long long i = (long long)blockIdx.x * blockDim.x + threadIdx.x;
    if (i < total) dst[i] = f2bf(src[i]);
}

// Transpose + convert: src (nb, rows, cols) -> dst (nb, cols, rows). Used for x^T.
__global__ void xpose_f32_to_bf16(const float* __restrict__ src,
                                  unsigned short* __restrict__ dst,
                                  int rows, int cols, long long total) {
    long long i = (long long)blockIdx.x * blockDim.x + threadIdx.x;
    if (i >= total) return;
    int c = (int)(i % cols);
    long long t = i / cols;
    int r = (int)(t % rows);
    long long z = t / rows;
    dst[(z * cols + c) * (long long)rows + r] = f2bf(src[i]);
}

// ---------------------------------------------------------------------------
// Generic batched WMMA GEMM: D = epilogue(A @ Bt^T)
//   A  (M x K, ld=ldA)  bf16 row-major
//   Bt (N x K, ld=ldBt) bf16 row-major (B transposed, K contiguous)
// Block tile 128(M) x 128(N), K-step 32. 8 waves: 4 row-waves x 2 col-waves,
// each wave computing 32x64 as 2x4 accumulators (B frags reused over M).
// Modes: 0 = +bias[col] -> bf16          (q/k projections; weights per z%H)
//        1 = +bias[row] -> bf16          (v projection; A per z%H, Bt per z/H)
//        2 = gamma*acc + x -> bf16       (attention output, heads residual)
//        3 = relu(acc+bias[col]) -> bf16 (FC1)
//        4 = relu(acc+bias[col]) + x -> f32 scatter (FC2 + final residual)
// ---------------------------------------------------------------------------
struct GemmArgs {
    const unsigned short* A;
    const unsigned short* Bt;
    unsigned short* Dbf;
    const float* bias;
    const float* gamma;
    const float* xOrig;   // original f32 x (residuals, modes 2 & 4)
    float* outF;          // final f32 output (mode 4)
    long long aBatch, bBatch, dBatch;
    int M, N, K, ldA, ldBt, ldD;
    int H, T, mode;
};

#define BM 128
#define BN 128
#define BK 32
#define LDS_K 40   // padded K-stride (ushorts): 80B rows, 16B aligned, bank-spread

__global__ __launch_bounds__(256) void gemm_wmma(GemmArgs g) {
    __shared__ alignas(16) unsigned short sA[2][BM * LDS_K];  // 2 x 10 KB [m][k]
    __shared__ alignas(16) unsigned short sB[2][BN * LDS_K];  // 2 x 10 KB [n][k]

    const int tid  = threadIdx.x;
    const int lane = tid & 31;
    const int wid  = tid >> 5;
    const int wm   = wid & 3;       // wave row (32 rows each)
    const int wn   = wid >> 2;      // wave col (64 cols each)
    const int z    = blockIdx.z;
    const int m0   = blockIdx.y * BM;
    const int n0   = blockIdx.x * BN;
    const int hi   = lane >> 4;     // lane half select
    const int r16  = lane & 15;

    const unsigned short* Ab = g.A;
    const unsigned short* Bb = g.Bt;
    const float* bias = g.bias;
    if (g.mode == 0) {
        Ab   += (long long)(z / g.H) * g.aBatch;   // x^T per batch b
        Bb   += (long long)(z % g.H) * g.bBatch;   // weights per head h
        bias += (long long)(z % g.H) * g.N;
    } else if (g.mode == 1) {
        Ab   += (long long)(z % g.H) * g.aBatch;   // Wv per head h
        Bb   += (long long)(z / g.H) * g.bBatch;   // x^T per batch b
        bias += (long long)(z % g.H) * g.M;        // bv indexed by row c
    } else if (g.mode == 2) {
        Ab += (long long)z * g.aBatch;
        Bb += (long long)z * g.bBatch;
    }

    f32x8 acc[2][4];
    #pragma unroll
    for (int mt = 0; mt < 2; ++mt)
        #pragma unroll
        for (int nt = 0; nt < 4; ++nt)
            acc[mt][nt] = (f32x8){0, 0, 0, 0, 0, 0, 0, 0};

    // Staging: A and B tiles are both 128 rows x 32 K; 2 threads/row, 32B each.
    const int srow = tid >> 1, sch = (tid & 1) * 2;

    // Per-thread async DMA of one K-step tile pair into LDS buffer `buf`.
    auto stage = [&](int buf, int k0) {
        const unsigned short* srcA = Ab + (size_t)(m0 + srow) * g.ldA + k0 + sch * 8;
        async_load_b128(lds_off(&sA[buf][srow * LDS_K + sch * 8]),       srcA);
        async_load_b128(lds_off(&sA[buf][srow * LDS_K + (sch + 1) * 8]), srcA + 8);
        const unsigned short* srcB = Bb + (size_t)(n0 + srow) * g.ldBt + k0 + sch * 8;
        async_load_b128(lds_off(&sB[buf][srow * LDS_K + sch * 8]),       srcB);
        async_load_b128(lds_off(&sB[buf][srow * LDS_K + (sch + 1) * 8]), srcB + 8);
    };

    // Prologue: fill buffer 0.
    stage(0, 0);
    wait_async0();
    __syncthreads();

    int buf = 0;
    for (int k0 = 0; k0 < g.K; k0 += BK) {
        // Kick off the DMA for the next tile into the other buffer; it runs
        // concurrently with this tile's WMMAs.
        if (k0 + BK < g.K) {
            stage(buf ^ 1, k0 + BK);
            if (k0 + 2 * BK < g.K) {   // warm L2 for the tile after next
                __builtin_prefetch(Ab + (size_t)(m0 + srow) * g.ldA  + k0 + 2 * BK, 0, 1);
                __builtin_prefetch(Bb + (size_t)(n0 + srow) * g.ldBt + k0 + 2 * BK, 0, 1);
            }
        }

        // A fragments (2 x 16x32): lanes 0-15 K{0-7,16-23}, lanes 16-31 K{8-15,24-31}.
        FragU af[2];
        #pragma unroll
        for (int mt = 0; mt < 2; ++mt) {
            const unsigned short* base = &sA[buf][(wm * 32 + mt * 16 + r16) * LDS_K];
            const int c0 = hi * 8;
            af[mt].q[0] = *(const u32x4*)(base + c0);
            af[mt].q[1] = *(const u32x4*)(base + c0 + 16);
        }
        #pragma unroll
        for (int nt = 0; nt < 4; ++nt) {
            // B fragment (32x16): lanes 0-15 K{0-7,16-23}, lanes 16-31 K{8-15,24-31}; reused over M.
            FragU bfr;
            const unsigned short* base =
                &sB[buf][(wn * 64 + nt * 16 + r16) * LDS_K + hi * 8];
            bfr.q[0] = *(const u32x4*)(base);
            bfr.q[1] = *(const u32x4*)(base + 16);
            #pragma unroll
            for (int mt = 0; mt < 2; ++mt)
                acc[mt][nt] = __builtin_amdgcn_wmma_f32_16x16x32_bf16(
                    false, af[mt].v, false, bfr.v, (short)0, acc[mt][nt], false, false);
        }

        // Drain this wave's outstanding DMAs, then make them block-visible.
        wait_async0();
        __syncthreads();
        buf ^= 1;
    }

    // Epilogue. C/D layout: VGPR r -> row (r + 8*hi), lane -> col.
    #pragma unroll
    for (int mt = 0; mt < 2; ++mt) {
        #pragma unroll
        for (int nt = 0; nt < 4; ++nt) {
            #pragma unroll
            for (int r = 0; r < 8; ++r) {
                const int row = m0 + wm * 32 + mt * 16 + r + hi * 8;
                const int col = n0 + wn * 64 + nt * 16 + r16;
                float v = acc[mt][nt][r];
                if (g.mode == 0) {
                    v += bias[col];
                    unsigned short* D = g.Dbf + (long long)z * g.dBatch;
                    D[(size_t)row * g.ldD + col] = f2bf(v);
                } else if (g.mode == 1) {
                    v += bias[row];
                    unsigned short* D = g.Dbf + (long long)z * g.dBatch;
                    D[(size_t)row * g.ldD + col] = f2bf(v);
                } else if (g.mode == 2) {
                    const float gm = g.gamma[z % g.H];
                    // residual x[b, c=col, t=row], exact f32 read
                    const size_t xi = ((size_t)(z / g.H) * g.N + col) * (size_t)g.T + row;
                    unsigned short* D = g.Dbf + (long long)z * g.dBatch;
                    D[(size_t)row * g.ldD + col] = f2bf(gm * v + g.xOrig[xi]);
                } else if (g.mode == 3) {
                    v += bias[col];
                    v = v > 0.0f ? v : 0.0f;
                    g.Dbf[(size_t)row * g.ldD + col] = f2bf(v);
                } else { // mode 4: out[b, col*H+h, t] = relu(acc+b2) + x[...]
                    v += bias[col];
                    v = v > 0.0f ? v : 0.0f;
                    const int HT = g.H * g.T;
                    const int b  = row / HT;
                    const int h  = (row / g.T) % g.H;
                    const int t  = row % g.T;
                    const size_t oi = ((size_t)b * (g.N * g.H) + (size_t)col * g.H + h)
                                      * (size_t)g.T + t;
                    g.outF[oi] = v + g.xOrig[oi];
                }
            }
        }
    }
}

// ---------------------------------------------------------------------------
// Attention scores + softmax. One wave per 16 query rows per (b,h).
// E strip (16 x T) lives in 64 KB LDS; exact row softmax; P emitted bf16.
// q/k fragments load directly from global (both are K-contiguous per lane).
// ---------------------------------------------------------------------------
__global__ __launch_bounds__(32) void attn_softmax(
        const unsigned short* __restrict__ qT,   // (B*H, T, 128) bf16
        const unsigned short* __restrict__ kT,   // (B*H, T, 128) bf16
        unsigned short* __restrict__ P,          // (B*H, T, T)   bf16
        int T, int D) {
    __shared__ alignas(16) float Es[16 * 1024];  // 64 KB (of 320 KB WGP LDS)

    const int lane = threadIdx.x;
    const int hi   = lane >> 4;
    const int r16  = lane & 15;
    const int t0   = blockIdx.x * 16;
    const long long zoff = (long long)blockIdx.y * T * D;
    const unsigned short* q = qT + zoff;
    const unsigned short* k = kT + zoff;

    // q fragments for all 4 K-chunks stay resident across the s sweep.
    FragU aq[4];
    #pragma unroll
    for (int kc = 0; kc < 4; ++kc) {
        const unsigned short* base = q + (size_t)(t0 + r16) * D + kc * 32 + hi * 8;
        aq[kc].q[0] = *(const u32x4*)(base);
        aq[kc].q[1] = *(const u32x4*)(base + 16);
    }

    for (int s0 = 0; s0 < T; s0 += 16) {
        f32x8 acc = (f32x8){0, 0, 0, 0, 0, 0, 0, 0};
        #pragma unroll
        for (int kc = 0; kc < 4; ++kc) {
            FragU bk;
            const unsigned short* base = k + (size_t)(s0 + r16) * D + kc * 32 + hi * 8;
            bk.q[0] = *(const u32x4*)(base);
            bk.q[1] = *(const u32x4*)(base + 16);
            acc = __builtin_amdgcn_wmma_f32_16x16x32_bf16(
                false, aq[kc].v, false, bk.v, (short)0, acc, false, false);
        }
        #pragma unroll
        for (int r = 0; r < 8; ++r)
            Es[(r + hi * 8) * 1024 + s0 + r16] = acc[r];
    }
    __syncthreads();

    unsigned short* Pz = P + (long long)blockIdx.y * T * T;
    for (int r = 0; r < 16; ++r) {
        float* E = &Es[r * 1024];
        float mx = -3.0e38f;
        for (int i = lane; i < T; i += 32) mx = fmaxf(mx, E[i]);
        #pragma unroll
        for (int o = 16; o > 0; o >>= 1) mx = fmaxf(mx, __shfl_xor(mx, o, 32));
        float sum = 0.0f;
        for (int i = lane; i < T; i += 32) {
            float e = __expf(E[i] - mx);
            E[i] = e;
            sum += e;
        }
        #pragma unroll
        for (int o = 16; o > 0; o >>= 1) sum += __shfl_xor(sum, o, 32);
        const float inv = 1.0f / sum;
        unsigned short* pr = Pz + (size_t)(t0 + r) * T;
        for (int i = lane; i < T; i += 32) pr[i] = f2bf(E[i] * inv);
    }
}

// ---------------------------------------------------------------------------
// Host-side orchestration.
// ---------------------------------------------------------------------------
extern "C" void kernel_launch(void* const* d_in, const int* in_sizes, int n_in,
                              void* d_out, int out_size, void* d_ws, size_t ws_size,
                              hipStream_t stream) {
    (void)in_sizes; (void)n_in; (void)out_size; (void)ws_size;
    const float* x     = (const float*)d_in[0];
    const float* Wq    = (const float*)d_in[1];
    const float* bq    = (const float*)d_in[2];
    const float* Wk    = (const float*)d_in[3];
    const float* bk    = (const float*)d_in[4];
    const float* Wv    = (const float*)d_in[5];
    const float* bv    = (const float*)d_in[6];
    const float* gamma = (const float*)d_in[7];
    const float* W1    = (const float*)d_in[8];
    const float* b1    = (const float*)d_in[9];
    const float* W2    = (const float*)d_in[10];
    const float* b2    = (const float*)d_in[11];
    float* out = (float*)d_out;

    constexpr int B = 8, C = 1024, T = 1024, H = 8, DQK = 128, DFC1 = 128;

    // Workspace carve-out (all bf16, 256B aligned).
    char* wp = (char*)d_ws;
    auto carve = [&](size_t bytes) {
        char* p = wp; wp += (bytes + 255) & ~(size_t)255; return (unsigned short*)p;
    };
    unsigned short* xT  = carve((size_t)B * T * C * 2);        //  16 MB  x^T (B,T,C)
    unsigned short* Wqb = carve((size_t)H * DQK * C * 2);      //   2 MB  (N,K) layout
    unsigned short* Wkb = carve((size_t)H * DQK * C * 2);      //   2 MB
    unsigned short* Wvb = carve((size_t)H * C * C * 2);        //  16 MB
    unsigned short* W1b = carve((size_t)DFC1 * C * 2);         // 256 KB
    unsigned short* W2b = carve((size_t)DFC1 * DFC1 * 2);      //  32 KB
    unsigned short* qT  = carve((size_t)B * H * T * DQK * 2);  //  16 MB  (z,t,o)
    unsigned short* kT  = carve((size_t)B * H * T * DQK * 2);  //  16 MB
    unsigned short* v   = carve((size_t)B * H * C * T * 2);    // 128 MB  (z,c,s)
    unsigned short* P   = carve((size_t)B * H * T * T * 2);    // 128 MB  (z,t,s)
    unsigned short* Yb  = carve((size_t)B * H * T * C * 2);    // 128 MB  heads (z,t,c)
    unsigned short* Z1  = carve((size_t)B * H * T * DFC1 * 2); //  16 MB

    // ---- Stage 0: bf16 copies. Only x is transposed; weights keep (N,K). ----
    {
        long long n = (long long)B * C * T;
        xpose_f32_to_bf16<<<dim3((unsigned)((n + 255) / 256)), 256, 0, stream>>>(x, xT, C, T, n);
        n = (long long)H * DQK * C;
        conv_f32_to_bf16<<<dim3((unsigned)((n + 255) / 256)), 256, 0, stream>>>(Wq, Wqb, n);
        conv_f32_to_bf16<<<dim3((unsigned)((n + 255) / 256)), 256, 0, stream>>>(Wk, Wkb, n);
        n = (long long)H * C * C;
        conv_f32_to_bf16<<<dim3((unsigned)((n + 255) / 256)), 256, 0, stream>>>(Wv, Wvb, n);
        n = (long long)DFC1 * C;
        conv_f32_to_bf16<<<dim3((unsigned)((n + 255) / 256)), 256, 0, stream>>>(W1, W1b, n);
        n = (long long)DFC1 * DFC1;
        conv_f32_to_bf16<<<dim3((unsigned)((n + 255) / 256)), 256, 0, stream>>>(W2, W2b, n);
    }

    GemmArgs g{};
    g.H = H; g.T = T;

    // ---- q^T = x^T @ Wq^T + bq :  A=x^T (T,C), Bt=Wq (DQK,C) ----
    g.A = xT;  g.aBatch = (long long)T * C;
    g.Bt = Wqb; g.bBatch = (long long)DQK * C;
    g.Dbf = qT; g.dBatch = (long long)T * DQK;
    g.bias = bq; g.M = T; g.N = DQK; g.K = C; g.ldA = C; g.ldBt = C; g.ldD = DQK;
    g.mode = 0;
    gemm_wmma<<<dim3(DQK / BN, T / BM, B * H), 256, 0, stream>>>(g);

    // ---- k^T ----
    g.Bt = Wkb; g.Dbf = kT; g.bias = bk;
    gemm_wmma<<<dim3(DQK / BN, T / BM, B * H), 256, 0, stream>>>(g);

    // ---- v = Wv @ x + bv :  A=Wv (C,C), Bt=x^T (T,C); D=(z,c,s) ld T ----
    g.A = Wvb; g.aBatch = (long long)C * C;
    g.Bt = xT; g.bBatch = (long long)T * C;
    g.Dbf = v; g.dBatch = (long long)C * T;
    g.bias = bv; g.M = C; g.N = T; g.K = C; g.ldA = C; g.ldBt = C; g.ldD = T;
    g.mode = 1;
    gemm_wmma<<<dim3(T / BN, C / BM, B * H), 256, 0, stream>>>(g);

    // ---- P = softmax(q^T k) per (b,h) ----
    attn_softmax<<<dim3(T / 16, B * H), 32, 0, stream>>>(qT, kT, P, T, DQK);

    // ---- heads: Y = gamma * (P @ v^T) + x :  A=P (T,T), Bt=v (C,T) ----
    g.A = P;  g.aBatch = (long long)T * T;
    g.Bt = v; g.bBatch = (long long)C * T;
    g.Dbf = Yb; g.dBatch = (long long)T * C;
    g.gamma = gamma; g.xOrig = x;
    g.M = T; g.N = C; g.K = T; g.ldA = T; g.ldBt = T; g.ldD = C;
    g.mode = 2;
    gemm_wmma<<<dim3(C / BN, T / BM, B * H), 256, 0, stream>>>(g);

    // ---- FC1: Z1 = relu(Y @ W1^T + b1) :  Bt=W1 (128,1024), M=65536 ----
    g.A = Yb; g.aBatch = 0;
    g.Bt = W1b; g.bBatch = 0;
    g.Dbf = Z1; g.dBatch = 0;
    g.bias = b1;
    g.M = B * H * T; g.N = DFC1; g.K = C; g.ldA = C; g.ldBt = C; g.ldD = DFC1;
    g.mode = 3;
    gemm_wmma<<<dim3(DFC1 / BN, (B * H * T) / BM, 1), 256, 0, stream>>>(g);

    // ---- FC2: out[b, d*H+h, t] = relu(Z1 @ W2^T + b2) + x ----
    g.A = Z1; g.Bt = W2b;
    g.bias = b2; g.outF = out; g.xOrig = x;
    g.M = B * H * T; g.N = DFC1; g.K = DFC1; g.ldA = DFC1; g.ldBt = DFC1; g.ldD = DFC1;
    g.mode = 4;
    gemm_wmma<<<dim3(DFC1 / BN, (B * H * T) / BM, 1), 256, 0, stream>>>(g);
}
